// Generator_66571993088085
// MI455X (gfx1250) — compile-verified, matrix operand order checked by reading, not run on hardware
//
#include <hip/hip_runtime.h>
#include <hip/hip_bf16.h>

// Problem constants from the reference
#define Bn 128
#define Tn 512
#define Vn 128
#define En 256
#define Hn 1024
#define Gn 4096   // 4*H

#define APAD 16                 // pad h-tile rows to 2080B (32B multiple)
#define AROW (Hn + APAD)        // elements per padded LDS row
#define ZROW 68                 // padded z row stride (floats)

typedef __attribute__((ext_vector_type(16))) __bf16 v16bf;
typedef __attribute__((ext_vector_type(8)))  float  v8f;

__device__ __forceinline__ unsigned short f32_to_bf16_bits(float f) {
    unsigned int u = __float_as_uint(f);
    unsigned int r = u + 0x7FFFu + ((u >> 16) & 1u);   // round-to-nearest-even
    return (unsigned short)(r >> 16);
}
__device__ __forceinline__ float bf16_bits_to_f32(unsigned short h) {
    return __uint_as_float(((unsigned int)h) << 16);
}
__device__ __forceinline__ float sigmoidf_(float x) {
    return 1.0f / (1.0f + __expf(-x));
}

// Async-stage one 16-row h tile (16 x Hn bf16, row-padded) into LDS.
// 2048 16-byte chunks, spread over all block threads; each lane issues
// global_load_async_to_lds_b128 (ASYNCcnt), then waits + barrier.
__device__ __forceinline__ void stage_h_tile_async(unsigned short* aTile,
                                                   const unsigned short* __restrict__ h_in,
                                                   int m0, int tid, int nthreads) {
    unsigned lds_base = (unsigned)(size_t)aTile;
    for (int i = tid; i < 2048; i += nthreads) {
        int row = i >> 7;                 // 128 chunks (16B) per 2048B row
        int cb  = (i & 127) << 4;         // byte offset within row
        unsigned laddr = lds_base + (unsigned)(row * (AROW * 2) + cb);
        unsigned long long gaddr =
            (unsigned long long)(size_t)((const char*)(h_in + (size_t)(m0 + row) * Hn) + cb);
        asm volatile("global_load_async_to_lds_b128 %0, %1, off"
                     :: "v"(laddr), "v"(gaddr) : "memory");
    }
    asm volatile("s_wait_asynccnt 0" ::: "memory");
    __syncthreads();
}

// ---------------------------------------------------------------------------
// Precompute: table[v, f] = sum_e emb[v,e] * W[e,f] + b[f]   (V x 4H, f32)
// ---------------------------------------------------------------------------
__global__ void precompute_table_kernel(const float* __restrict__ emb,
                                        const float* __restrict__ W,
                                        const float* __restrict__ bvec,
                                        float* __restrict__ table) {
    int idx = blockIdx.x * blockDim.x + threadIdx.x;   // V*4H = 524288
    if (idx >= Vn * Gn) return;
    int v = idx / Gn;
    int f = idx - v * Gn;
    float acc = bvec[f];
    const float* er = emb + v * En;
#pragma unroll 4
    for (int e = 0; e < En; ++e)
        acc = fmaf(er[e], W[e * Gn + f], acc);
    table[idx] = acc;
}

// U (H x 4H, f32) -> UT (4H x H, bf16)
__global__ void transpose_U_kernel(const float* __restrict__ U,
                                   unsigned short* __restrict__ UT) {
    int idx = blockIdx.x * blockDim.x + threadIdx.x;   // 4H*H
    if (idx >= Gn * Hn) return;
    int n = idx / Hn;           // z / gate column 0..4095
    int k = idx - n * Hn;       // hidden k 0..1023
    UT[idx] = f32_to_bf16_bits(U[k * Gn + n]);
}

// Wd (H x V, f32) -> WdT (V x H, bf16)
__global__ void transpose_Wd_kernel(const float* __restrict__ Wd,
                                    unsigned short* __restrict__ WdT) {
    int idx = blockIdx.x * blockDim.x + threadIdx.x;   // V*H
    if (idx >= Vn * Hn) return;
    int v = idx / Hn;
    int k = idx - v * Hn;
    WdT[idx] = f32_to_bf16_bits(Wd[k * Vn + v]);
}

// zero h0 (bf16 bits of 0.0f == 0) and c0
__global__ void init_state_kernel(unsigned short* __restrict__ h0,
                                  float* __restrict__ c0) {
    int idx = blockIdx.x * blockDim.x + threadIdx.x;   // B*H
    if (idx >= Bn * Hn) return;
    h0[idx] = 0;
    c0[idx] = 0.0f;
}

// ---------------------------------------------------------------------------
// One LSTM timestep.
// grid (B/16, H/64), block 256 threads = 8 waves.
// Wave w owns gate (w>>1), n-half (w&1): a 16x32 tile of z = h_old @ U.
// A tile (h rows) async-staged in LDS; B from global with register
// double-buffering so loads overlap WMMAs. Gate fusion via LDS.
// ---------------------------------------------------------------------------
__global__ void __launch_bounds__(256)
lstm_step_kernel(int t,
                 const int* __restrict__ tokens,
                 const float* __restrict__ table,        // V x 4H (has bias)
                 const unsigned short* __restrict__ UT,  // 4H x H bf16
                 const unsigned short* __restrict__ h_in,  // B x H bf16
                 unsigned short* __restrict__ h_out,       // B x H bf16
                 float* __restrict__ c) {                  // B x H f32
    const int m0     = blockIdx.x * 16;     // batch tile
    const int n_base = blockIdx.y * 64;     // hidden-col tile
    const int tid    = threadIdx.x;
    const int wave   = tid >> 5;            // 0..7
    const int lane   = tid & 31;
    const int lrow   = lane & 15;           // M / N sub-index
    const int lhalf  = lane >> 4;           // K half select
    const int gate   = wave >> 1;           // 0..3 (i,f,g,o)
    const int nh     = (wave & 1) * 32;     // n-half within block tile

    __shared__ __align__(32) unsigned short aTile[16 * AROW];  // ~32.5 KB
    __shared__ float zlds[4 * 16 * ZROW];                      // ~17 KB

    stage_h_tile_async(aTile, h_in, m0, tid, 256);

    // Operand bases.
    const unsigned short* ar = aTile + lrow * AROW + lhalf * 16;   // LDS
    const int col0 = gate * Hn + n_base + nh + 0 * 16 + lrow;      // z cols
    const int col1 = gate * Hn + n_base + nh + 1 * 16 + lrow;
    const unsigned short* br0 = UT + (size_t)col0 * Hn + lhalf * 16;
    const unsigned short* br1 = UT + (size_t)col1 * Hn + lhalf * 16;

    v8f acc0 = v8f{0}, acc1 = v8f{0};

    // Software-pipelined K loop: prefetch k+32 while doing WMMAs on k.
    v16bf a_cur  = *(const v16bf*)(ar);
    v16bf b0_cur = *(const v16bf*)(br0);
    v16bf b1_cur = *(const v16bf*)(br1);
    for (int k = 0; k < Hn; k += 32) {
        int kn = (k + 32) & (Hn - 1);
        v16bf a_nxt  = *(const v16bf*)(ar + kn);
        v16bf b0_nxt = *(const v16bf*)(br0 + kn);
        v16bf b1_nxt = *(const v16bf*)(br1 + kn);
        acc0 = __builtin_amdgcn_wmma_f32_16x16x32_bf16(
            false, a_cur, false, b0_cur, (short)0, acc0, false, false);
        acc1 = __builtin_amdgcn_wmma_f32_16x16x32_bf16(
            false, a_cur, false, b1_cur, (short)0, acc1, false, false);
        a_cur = a_nxt; b0_cur = b0_nxt; b1_cur = b1_nxt;
    }

    // Spill z tiles to LDS per documented C/D layout:
    // VGPR r, lanes 0-15 -> M=r, lanes 16-31 -> M=r+8; N = lane%16.
#pragma unroll
    for (int r = 0; r < 8; ++r) {
        int m = r + 8 * lhalf;
        zlds[gate * (16 * ZROW) + m * ZROW + nh + 0 * 16 + lrow] = acc0[r];
        zlds[gate * (16 * ZROW) + m * ZROW + nh + 1 * 16 + lrow] = acc1[r];
    }
    __syncthreads();

    // Elementwise gate fusion: 1024 elements, 256 threads -> 4 each.
    for (int e = tid; e < 16 * 64; e += 256) {
        int m = e >> 6;
        int n = e & 63;
        int b = m0 + m;
        int hc = n_base + n;
        int tok = tokens[b * Tn + t];
        const float* tr = table + (size_t)tok * Gn + hc;
        const float* zr = zlds + m * ZROW + n;
        float zi = zr[0 * (16 * ZROW)] + tr[0 * Hn];
        float zf = zr[1 * (16 * ZROW)] + tr[1 * Hn];
        float zg = zr[2 * (16 * ZROW)] + tr[2 * Hn];
        float zo = zr[3 * (16 * ZROW)] + tr[3 * Hn];
        float ig = sigmoidf_(zi);
        float fg = sigmoidf_(zf);
        float gg = tanhf(zg);
        float og = sigmoidf_(zo);
        float c_old = c[b * Hn + hc];
        float h_old = bf16_bits_to_f32(h_in[b * Hn + hc]);
        float c_new = fg * c_old + ig * gg;
        float h_new = og * tanhf(c_new);
        bool msk = (tok != 0);
        c[b * Hn + hc]     = msk ? c_new : c_old;
        h_out[b * Hn + hc] = f32_to_bf16_bits(msk ? h_new : h_old);
    }
}

// ---------------------------------------------------------------------------
// Decoder for one timestep: probs[:, t, :] = softmax(h @ Wd + bd).
// grid B/16, block 256 threads = 8 waves; wave w owns logit cols [16w,16w+16).
// A tile async-staged in LDS (shared by all 8 waves); B register-pipelined.
// ---------------------------------------------------------------------------
__global__ void __launch_bounds__(256)
decode_step_kernel(int t,
                   const unsigned short* __restrict__ h,   // B x H bf16
                   const unsigned short* __restrict__ WdT, // V x H bf16
                   const float* __restrict__ bd,
                   float* __restrict__ out) {              // B x T x V
    const int m0    = blockIdx.x * 16;
    const int tid   = threadIdx.x;
    const int wave  = tid >> 5;
    const int lane  = tid & 31;
    const int lrow  = lane & 15;
    const int lhalf = lane >> 4;

    __shared__ __align__(32) unsigned short aTile[16 * AROW];  // ~32.5 KB
    __shared__ float lg[16 * 128];                             // logits tile

    stage_h_tile_async(aTile, h, m0, tid, 256);

    const unsigned short* ar = aTile + lrow * AROW + lhalf * 16;
    const unsigned short* br = WdT + (size_t)(wave * 16 + lrow) * Hn + lhalf * 16;

    v8f acc = v8f{0};
    v16bf a_cur = *(const v16bf*)(ar);
    v16bf b_cur = *(const v16bf*)(br);
    for (int k = 0; k < Hn; k += 32) {
        int kn = (k + 32) & (Hn - 1);
        v16bf a_nxt = *(const v16bf*)(ar + kn);
        v16bf b_nxt = *(const v16bf*)(br + kn);
        acc = __builtin_amdgcn_wmma_f32_16x16x32_bf16(
            false, a_cur, false, b_cur, (short)0, acc, false, false);
        a_cur = a_nxt; b_cur = b_nxt;
    }

    float bias = bd[wave * 16 + lrow];
#pragma unroll
    for (int r = 0; r < 8; ++r) {
        int m = r + 8 * lhalf;
        lg[m * 128 + wave * 16 + lrow] = acc[r] + bias;
    }
    __syncthreads();

    // Row softmax: one thread per row (16 rows).
    if (tid < 16) {
        int m = tid;
        int b = m0 + m;
        float mx = -3.4028235e38f;
        for (int v = 0; v < Vn; ++v) mx = fmaxf(mx, lg[m * 128 + v]);
        float s = 0.0f;
        for (int v = 0; v < Vn; ++v) {
            float e = __expf(lg[m * 128 + v] - mx);
            lg[m * 128 + v] = e;
            s += e;
        }
        float inv = 1.0f / s;
        float* orow = out + (size_t)(b * Tn + t) * Vn;
        for (int v = 0; v < Vn; ++v) orow[v] = lg[m * 128 + v] * inv;
    }
}

// ---------------------------------------------------------------------------
extern "C" void kernel_launch(void* const* d_in, const int* in_sizes, int n_in,
                              void* d_out, int out_size, void* d_ws, size_t ws_size,
                              hipStream_t stream) {
    (void)in_sizes; (void)n_in; (void)out_size; (void)ws_size;
    const int*   tokens = (const int*)d_in[0];
    const float* emb    = (const float*)d_in[1];
    const float* W      = (const float*)d_in[2];
    const float* U      = (const float*)d_in[3];
    const float* bvec   = (const float*)d_in[4];
    const float* Wd     = (const float*)d_in[5];
    const float* bd     = (const float*)d_in[6];
    float* out = (float*)d_out;

    // Workspace carve-up (all offsets 256B-aligned).
    char* ws = (char*)d_ws;
    float*          table = (float*)ws;          ws += (size_t)Vn * Gn * sizeof(float);          // 2 MB
    unsigned short* UT    = (unsigned short*)ws; ws += (size_t)Gn * Hn * sizeof(unsigned short); // 8 MB
    unsigned short* WdT   = (unsigned short*)ws; ws += (size_t)Vn * Hn * sizeof(unsigned short); // 256 KB
    unsigned short* hA    = (unsigned short*)ws; ws += (size_t)Bn * Hn * sizeof(unsigned short); // 256 KB
    unsigned short* hB    = (unsigned short*)ws; ws += (size_t)Bn * Hn * sizeof(unsigned short); // 256 KB
    float*          cbuf  = (float*)ws;          ws += (size_t)Bn * Hn * sizeof(float);          // 512 KB

    // One-time precompute (per call; deterministic).
    precompute_table_kernel<<<(Vn * Gn + 255) / 256, 256, 0, stream>>>(emb, W, bvec, table);
    transpose_U_kernel<<<(Gn * Hn + 255) / 256, 256, 0, stream>>>(U, UT);
    transpose_Wd_kernel<<<(Vn * Hn + 255) / 256, 256, 0, stream>>>(Wd, WdT);
    init_state_kernel<<<(Bn * Hn + 255) / 256, 256, 0, stream>>>(hA, cbuf);

    unsigned short* hbuf[2] = {hA, hB};
    for (int t = 0; t < Tn; ++t) {
        unsigned short* hin  = hbuf[t & 1];
        unsigned short* hout = hbuf[(t + 1) & 1];
        lstm_step_kernel<<<dim3(Bn / 16, Hn / 64), 256, 0, stream>>>(
            t, tokens, table, UT, hin, hout, cbuf);
        decode_step_kernel<<<Bn / 16, 256, 0, stream>>>(t, hout, WdT, bd, out);
    }
}
